// MultiStreamGAT_84404697301690
// MI455X (gfx1250) — matrix operand order repaired and checked by reading, not compiled, not run on hardware
//
#include <hip/hip_runtime.h>

// ---------------------------------------------------------------------------
// MultiStreamGAT for gfx1250 (MI455X).
// Pipeline (all on `stream`, graph-capture safe):
//   K0 : zero BN stat accumulators; convert W2/gate_w/ow to f16 in workspace
//   K1 : per-graph GAT layer 1 (F_in=3, scalar) -> pre-BN h1 (f16) + BN1 stats
//   K2 : BN1 finalize (scale/bias)
//   K3 : per-graph GAT layer 2: TDM (tensor_load_to_lds) DMAs W2(f16,128KB)
//        into LDS while BN1+ELU builds the f16 A tile; WMMA A@W2 with B
//        fragments fed from LDS; attention; head-mean -> pre-BN h2 + BN2 stats
//   K4 : BN2 finalize
//   K5 : node-mean -> pool linear -> LayerNorm -> ReLU -> s_out (BT,256)
//   K6 : fusion: WMMA gate GEMM (768x768), sigmoid*cat, WMMA out GEMM
//        (768x256), row LayerNorm + ReLU -> d_out
//
// Input mapping assumption: setup_inputs() flattened in dict-insertion order:
//   [0]=x, [1]=hand_adj, [2]=face_adj, [3]=body_adj,
//   then per stream s in (hand,face,body), base 4+12s:
//     +0 W1, +1 a1, +2 g1, +3 b1, +4 W2, +5 a2, +6 g2, +7 b2,
//     +8 pw, +9 pb, +10 lg, +11 lb
//   [40]=gate_w, [41]=gate_b, [42]=ow, [43]=ob, [44]=og, [45]=obeta
// ---------------------------------------------------------------------------

typedef __attribute__((ext_vector_type(16))) _Float16     v16h;
typedef __attribute__((ext_vector_type(8)))  float        v8f;
typedef __attribute__((ext_vector_type(4)))  unsigned int v4u;
typedef __attribute__((ext_vector_type(8)))  int          v8i_t;
typedef __attribute__((ext_vector_type(4)))  int          v4i_t;

#define GAT_EPS   1e-5f
#define GAT_LEAKY 0.2f
#define BT_TOT    4096      // B*T = 32*128
#define NMAX      48        // max N padded to 16

// ---- WMMA fragment helpers (CDNA5 ISA 7.12.2 layouts) ----------------------

// A-matrix 16x32 f16: lanes 0-15 hold rows M=0..15 (K=0..7,16..23),
// lanes 16-31 hold the same rows (K=8..15,24..31); VGPR packs K pairs.
__device__ static inline v16h load_a_frag(const _Float16* __restrict__ A,
                                          int lda, int lane) {
  const int hl = (lane >> 4) & 1;
  const int r  = lane & 15;
  v16h a;
#pragma unroll
  for (int v = 0; v < 8; ++v) {
    const int kb = ((v < 4) ? 0 : 16) + (hl ? 8 : 0) + 2 * (v & 3);
    a[2 * v]     = A[r * lda + kb];
    a[2 * v + 1] = A[r * lda + kb + 1];
  }
  return a;
}

// B-matrix 32x16 f16 (K x N row-major in memory): lane L holds column
// N=L&15, K order matches A (K=0..7,16..23 / 8..15,24..31 per lane half).
__device__ static inline v16h load_b_frag(const _Float16* __restrict__ B,
                                          int ldb, int lane) {
  const int hl = (lane >> 4) & 1;
  const _Float16* col = B + (lane & 15);
  v16h b;
#pragma unroll
  for (int j = 0; j < 16; ++j) {
    const int k = ((j < 8) ? 0 : 16) + (hl ? 8 : 0) + (j & 7);
    b[j] = col[(size_t)k * ldb];
  }
  return b;
}

// C/D 16x16 f32: VGPR v -> M = v + 8*(lane>=16), N = lane&15
#define CD_ROW(v, lane) ((v) + 8 * (((lane) >> 4) & 1))
#define CD_COL(lane)    ((lane) & 15)

__device__ static inline float elu_f(float v) {
  return v > 0.f ? v : __expf(v) - 1.f;
}

// ---- Tensor Data Mover: DMA a 2D f16 tensor (full-tile) into LDS -----------
// Builds a D# per CDNA5 ISA 8.3/8.4 and issues TENSOR_LOAD_TO_LDS.
// 5-arg builtin on ROCm 7.2 (clang-22), 6-arg on clang-23 toolchains.
#if defined(__has_builtin)
#if __has_builtin(__builtin_amdgcn_tensor_load_to_lds)
#define GAT_HAVE_TDM 1
#endif
#endif

#ifdef GAT_HAVE_TDM
__device__ static inline void tdm_load_2d_f16(const _Float16* gptr,
                                              unsigned lds_off,
                                              int dim0, int dim1, int stride0) {
  const unsigned long long ga = (unsigned long long)(uintptr_t)gptr;
  v4u g0;
  g0[0] = 1u;                                    // count=1 (valid, user mode)
  g0[1] = lds_off;                               // lds_addr (bytes)
  g0[2] = (unsigned)(ga & 0xFFFFFFFFu);          // global_addr[31:0]
  g0[3] = (unsigned)((ga >> 32) & 0x1FFFFFFu)    // global_addr[56:32]
        | (2u << 30);                            // type=2 ("image")
  v8i_t g1;
  g1[0] = (1 << 16);                             // wg_mask=0, data_size=1 (2B)
  g1[1] = (dim0 & 0xFFFF) << 16;                 // tensor_dim0[15:0]
  g1[2] = ((dim0 >> 16) & 0xFFFF)                // tensor_dim0[31:16]
        | ((dim1 & 0xFFFF) << 16);               // tensor_dim1[15:0]
  g1[3] = ((dim1 >> 16) & 0xFFFF)                // tensor_dim1[31:16]
        | ((dim0 & 0xFFFF) << 16);               // tile_dim0 = dim0
  g1[4] = (dim1 & 0xFFFF);                       // tile_dim1 = dim1, tile_dim2=0
  g1[5] = stride0;                               // tensor_dim0_stride[31:0]
  g1[6] = 0;                                     // stride0[47:32]=0, dim1_stride lo=0
  g1[7] = 0;
  v4i_t gz = {0, 0, 0, 0};
#if __clang_major__ >= 23
  v8i_t z8 = {};
  __builtin_amdgcn_tensor_load_to_lds(g0, g1, gz, gz, z8, 0);
#else
  __builtin_amdgcn_tensor_load_to_lds(g0, g1, gz, gz, 0);
#endif
}
#endif

// ---- K0 helpers ------------------------------------------------------------

__global__ void zero_f32_kernel(float* __restrict__ p, int n) {
  int i = blockIdx.x * blockDim.x + threadIdx.x;
  if (i < n) p[i] = 0.f;
}

__global__ void cvt_f32_to_f16_kernel(const float* __restrict__ s,
                                      _Float16* __restrict__ d, int n) {
  int i = blockIdx.x * blockDim.x + threadIdx.x;
  if (i < n) d[i] = (_Float16)s[i];
}

// ---- K1: GAT layer 1 (scalar; F_in = 3) ------------------------------------

__global__ void gat1_kernel(const float* __restrict__ x, int xoff, int xstride,
                            const float* __restrict__ adj, int N,
                            const float* __restrict__ W1,   // 3 x 256
                            const float* __restrict__ a1,   // 4 x 128
                            _Float16* __restrict__ h1out,   // BT*N*256 pre-BN
                            float* __restrict__ bn1sum,     // 256
                            float* __restrict__ bn1sq) {    // 256
  extern __shared__ char smem[];
  float* Wh    = (float*)smem;                       // 48*256
  float* alpha = Wh + NMAX * 256;                    // 4*48*48
  float* si    = alpha + 4 * NMAX * NMAX;            // 4*48
  float* sj    = si + 4 * NMAX;                      // 4*48
  float* hfeat = sj + 4 * NMAX;                      // 48*3

  const int tid = threadIdx.x;
  const int bt  = blockIdx.x;
  const float* xrow = x + (size_t)bt * xstride + xoff;

  for (int i = tid; i < N * 3; i += 256) hfeat[i] = xrow[i];
  __syncthreads();

  // Wh = h @ W1  (N x 256)
  for (int i = tid; i < N * 256; i += 256) {
    const int n = i >> 8, c = i & 255;
    Wh[i] = hfeat[n * 3 + 0] * W1[c] +
            hfeat[n * 3 + 1] * W1[256 + c] +
            hfeat[n * 3 + 2] * W1[512 + c];
  }
  __syncthreads();

  // si/sj per (head, node)
  for (int i = tid; i < 4 * N; i += 256) {
    const int h = i / N, n = i % N;
    const float* w  = Wh + n * 256 + h * 64;
    const float* ai = a1 + h * 128;
    float aI = 0.f, aJ = 0.f;
    for (int d = 0; d < 64; ++d) { aI += w[d] * ai[d]; aJ += w[d] * ai[64 + d]; }
    si[h * NMAX + n] = aI;
    sj[h * NMAX + n] = aJ;
  }
  __syncthreads();

  // masked leaky-softmax rows
  for (int i = tid; i < 4 * N; i += 256) {
    const int h = i / N, r = i % N;
    const float sir = si[h * NMAX + r];
    float m = -1e30f; int any = 0;
    for (int j = 0; j < N; ++j) {
      if (adj[r * N + j] != 0.f) {
        float e = sir + sj[h * NMAX + j];
        e = e > 0.f ? e : GAT_LEAKY * e;
        if (e > m) m = e;
        any = 1;
      }
    }
    float* arow = alpha + (h * NMAX + r) * NMAX;
    float s = 0.f;
    for (int j = 0; j < N; ++j) {
      float v = 0.f;
      if (any && adj[r * N + j] != 0.f) {
        float e = sir + sj[h * NMAX + j];
        e = e > 0.f ? e : GAT_LEAKY * e;
        v = __expf(e - m);
      }
      arow[j] = v;
      s += v;
    }
    const float inv = s > 0.f ? 1.f / s : 0.f;
    for (int j = 0; j < N; ++j) arow[j] *= inv;
  }
  __syncthreads();

  // h_new[n, h*64+d] = sum_j alpha[h,n,j] * Wh[j, h*64+d]  (concat heads)
  float sacc = 0.f, sqacc = 0.f;
  for (int i = tid; i < N * 256; i += 256) {        // col == tid (mod 256)
    const int n = i >> 8, c = i & 255, h = c >> 6;
    const float* arow = alpha + (h * NMAX + n) * NMAX;
    float acc = 0.f;
    for (int j = 0; j < N; ++j) acc += arow[j] * Wh[j * 256 + c];
    h1out[(size_t)bt * (size_t)(N * 256) + i] = (_Float16)acc;
    sacc += acc; sqacc += acc * acc;
  }
  atomicAdd(&bn1sum[tid], sacc);
  atomicAdd(&bn1sq[tid], sqacc);
}

// ---- K2/K4: BatchNorm finalize ---------------------------------------------

__global__ void bn_finalize_kernel(const float* __restrict__ sum,
                                   const float* __restrict__ sq,
                                   const float* __restrict__ g,
                                   const float* __restrict__ b,
                                   float* __restrict__ scale,
                                   float* __restrict__ bias,
                                   int C, float invM) {
  int c = blockIdx.x * blockDim.x + threadIdx.x;
  if (c < C) {
    const float mu  = sum[c] * invM;
    const float var = sq[c] * invM - mu * mu;
    const float s   = g[c] * rsqrtf(var + GAT_EPS);
    scale[c] = s;
    bias[c]  = b[c] - mu * s;
  }
}

// ---- K3: GAT layer 2 (TDM W2 -> LDS, WMMA for h @ W2) ----------------------

__global__ void gat2_kernel(const _Float16* __restrict__ h1,   // BT*N*256 pre-BN
                            const float* __restrict__ bn1s,
                            const float* __restrict__ bn1b,
                            const float* __restrict__ adj, int N, int Np,
                            const _Float16* __restrict__ W2h,  // 256 x 256 f16
                            const float* __restrict__ a2,      // 4 x 128
                            float* __restrict__ h2out,         // BT*N*64 pre-BN
                            float* __restrict__ bn2sum,        // 64
                            float* __restrict__ bn2sq) {       // 64
  extern __shared__ char smem[];
  float*    Wh2   = (float*)smem;                    // 48*256 f32
  float*    alpha = Wh2 + NMAX * 256;                // 4*48*48
  float*    si    = alpha + 4 * NMAX * NMAX;         // 4*48
  float*    sj    = si + 4 * NMAX;                   // 4*48
  _Float16* A     = (_Float16*)(sj + 4 * NMAX);      // 48*256 f16
  _Float16* Bl    = A + NMAX * 256;                  // 256*256 f16 (128 KB)

  const int tid  = threadIdx.x;
  const int lane = tid & 31;
  const int wv   = tid >> 5;
  const int bt   = blockIdx.x;

#ifdef GAT_HAVE_TDM
  // Kick the W2 DMA first so it overlaps the A-tile preparation.
  if (wv == 0)
    tdm_load_2d_f16(W2h, (unsigned)(uintptr_t)(void*)Bl, 256, 256, 256);
#endif

  // BN1 affine + ELU -> f16 A tile (rows >= N zero-padded)
  const _Float16* hrow = h1 + (size_t)bt * (size_t)(N * 256);
  for (int i = tid; i < Np * 256; i += 256) {
    const int n = i >> 8, c = i & 255;
    float v = 0.f;
    if (n < N) v = elu_f((float)hrow[n * 256 + c] * bn1s[c] + bn1b[c]);
    A[i] = (_Float16)v;
  }

#ifdef GAT_HAVE_TDM
  if (wv == 0) __builtin_amdgcn_s_wait_tensorcnt(0);
#else
  {  // fallback: cooperative global->LDS copy of W2 (128 KB, uint4)
    const uint4* srcv = (const uint4*)W2h;
    uint4* dstv = (uint4*)Bl;
    for (int i = tid; i < 8192; i += 256) dstv[i] = srcv[i];
  }
#endif
  __syncthreads();

  // Wh2 = A(Np x 256) @ W2(256 x 256)  via WMMA f32_16x16x32_f16, B from LDS
  const int mTiles = Np >> 4;
  const int nTiles = mTiles * 16;          // divisible by 8 waves (48/48/16)
  for (int t = wv; t < nTiles; t += 8) {
    const int mt = t >> 4, nt = t & 15;
    v8f acc = {};
    for (int k0 = 0; k0 < 256; k0 += 32) {
      v16h af = load_a_frag(A + mt * 16 * 256 + k0, 256, lane);
      v16h bf = load_b_frag(Bl + k0 * 256 + nt * 16, 256, lane);
      acc = __builtin_amdgcn_wmma_f32_16x16x32_f16(
          false, af, false, bf, (short)0, acc, false, false);
    }
#pragma unroll
    for (int v = 0; v < 8; ++v)
      Wh2[(mt * 16 + CD_ROW(v, lane)) * 256 + nt * 16 + CD_COL(lane)] = acc[v];
  }
  __syncthreads();

  // si/sj
  for (int i = tid; i < 4 * N; i += 256) {
    const int h = i / N, n = i % N;
    const float* w  = Wh2 + n * 256 + h * 64;
    const float* ai = a2 + h * 128;
    float aI = 0.f, aJ = 0.f;
    for (int d = 0; d < 64; ++d) { aI += w[d] * ai[d]; aJ += w[d] * ai[64 + d]; }
    si[h * NMAX + n] = aI;
    sj[h * NMAX + n] = aJ;
  }
  __syncthreads();

  // masked leaky-softmax
  for (int i = tid; i < 4 * N; i += 256) {
    const int h = i / N, r = i % N;
    const float sir = si[h * NMAX + r];
    float m = -1e30f; int any = 0;
    for (int j = 0; j < N; ++j) {
      if (adj[r * N + j] != 0.f) {
        float e = sir + sj[h * NMAX + j];
        e = e > 0.f ? e : GAT_LEAKY * e;
        if (e > m) m = e;
        any = 1;
      }
    }
    float* arow = alpha + (h * NMAX + r) * NMAX;
    float s = 0.f;
    for (int j = 0; j < N; ++j) {
      float v = 0.f;
      if (any && adj[r * N + j] != 0.f) {
        float e = sir + sj[h * NMAX + j];
        e = e > 0.f ? e : GAT_LEAKY * e;
        v = __expf(e - m);
      }
      arow[j] = v;
      s += v;
    }
    const float inv = s > 0.f ? 1.f / s : 0.f;
    for (int j = 0; j < N; ++j) arow[j] *= inv;
  }
  __syncthreads();

  // mean over heads: h2[n,d] = 0.25 * sum_h sum_j alpha[h,n,j]*Wh2[j,h*64+d]
  float sacc = 0.f, sqacc = 0.f;
  for (int i = tid; i < N * 64; i += 256) {         // col == tid (mod 64)
    const int n = i >> 6, d = i & 63;
    float acc = 0.f;
#pragma unroll
    for (int h = 0; h < 4; ++h) {
      const float* arow = alpha + (h * NMAX + n) * NMAX;
      const float* colp = Wh2 + h * 64 + d;
      for (int j = 0; j < N; ++j) acc += arow[j] * colp[j * 256];
    }
    acc *= 0.25f;
    h2out[(size_t)bt * (size_t)(N * 64) + i] = acc;
    sacc += acc; sqacc += acc * acc;
  }
  atomicAdd(&bn2sum[tid & 63], sacc);
  atomicAdd(&bn2sq[tid & 63], sqacc);
}

// ---- K5: node-mean -> pool linear -> LayerNorm -> ReLU ---------------------

__global__ void pool_kernel(const float* __restrict__ h2, int N,
                            const float* __restrict__ bn2s,
                            const float* __restrict__ bn2b,
                            const float* __restrict__ pw,   // 64 x 256
                            const float* __restrict__ pb,   // 256
                            const float* __restrict__ lg,
                            const float* __restrict__ lb,
                            float* __restrict__ sout) {     // BT x 256
  __shared__ float mvec[64];
  __shared__ float red1[256];
  __shared__ float red2[256];
  const int tid = threadIdx.x;
  const int bt  = blockIdx.x;
  const float* base = h2 + (size_t)bt * (size_t)(N * 64);

  if (tid < 64) {
    float acc = 0.f;
    for (int n = 0; n < N; ++n)
      acc += elu_f(base[n * 64 + tid] * bn2s[tid] + bn2b[tid]);
    mvec[tid] = acc / (float)N;
  }
  __syncthreads();

  float o = pb[tid];
  for (int d = 0; d < 64; ++d) o += mvec[d] * pw[d * 256 + tid];

  red1[tid] = o;
  red2[tid] = o * o;
  __syncthreads();
  for (int s = 128; s > 0; s >>= 1) {
    if (tid < s) { red1[tid] += red1[tid + s]; red2[tid] += red2[tid + s]; }
    __syncthreads();
  }
  const float mu  = red1[0] * (1.f / 256.f);
  const float var = red2[0] * (1.f / 256.f) - mu * mu;
  const float inv = rsqrtf(var + GAT_EPS);
  float v = (o - mu) * inv * lg[tid] + lb[tid];
  sout[(size_t)bt * 256 + tid] = v > 0.f ? v : 0.f;
}

// ---- K6: gated fusion (WMMA) + LayerNorm + ReLU ----------------------------

__global__ void fusion_kernel(const float* __restrict__ sh,
                              const float* __restrict__ sf,
                              const float* __restrict__ sb,
                              const _Float16* __restrict__ gwh,  // 768x768 f16
                              const float* __restrict__ gate_b,  // 768
                              const _Float16* __restrict__ owh,  // 768x256 f16
                              const float* __restrict__ ob,      // 256
                              const float* __restrict__ og,
                              const float* __restrict__ obeta,
                              float* __restrict__ out) {         // BT x 256
  extern __shared__ char smem[];
  _Float16* catL   = (_Float16*)smem;            // 16*768
  _Float16* fusedL = catL + 16 * 768;            // 16*768
  float*    outL   = (float*)(fusedL + 16 * 768);// 16*256
  float*    red    = outL + 16 * 256;            // 2*256

  const int tid  = threadIdx.x;
  const int lane = tid & 31;
  const int wv   = tid >> 5;
  const int row0 = blockIdx.x * 16;

  auto cat_val = [&](int bt, int c) -> float {
    return (c < 256) ? sh[(size_t)bt * 256 + c]
         : (c < 512) ? sf[(size_t)bt * 256 + (c - 256)]
                     : sb[(size_t)bt * 256 + (c - 512)];
  };

  for (int i = tid; i < 16 * 768; i += 256) {
    const int r = i / 768, c = i % 768;
    catL[i] = (_Float16)cat_val(row0 + r, c);
  }
  __syncthreads();

  // gate = cat(16x768) @ gate_w(768x768): 48 N-tiles, 6 per wave
  v8f accg[6];
#pragma unroll
  for (int i = 0; i < 6; ++i) accg[i] = (v8f){};
  for (int k0 = 0; k0 < 768; k0 += 32) {
    v16h af = load_a_frag(catL + k0, 768, lane);
#pragma unroll
    for (int i = 0; i < 6; ++i) {
      const int nt = wv * 6 + i;
      if (k0 + 32 < 768)
        __builtin_prefetch((const void*)(gwh + (size_t)(k0 + 32) * 768 + nt * 16), 0, 1);
      v16h bf = load_b_frag(gwh + (size_t)k0 * 768 + nt * 16, 768, lane);
      accg[i] = __builtin_amdgcn_wmma_f32_16x16x32_f16(
          false, af, false, bf, (short)0, accg[i], false, false);
    }
  }
#pragma unroll
  for (int i = 0; i < 6; ++i) {
    const int nt = wv * 6 + i;
#pragma unroll
    for (int v = 0; v < 8; ++v) {
      const int m = CD_ROW(v, lane), c = nt * 16 + CD_COL(lane);
      float g = accg[i][v] + gate_b[c];
      g = 1.f / (1.f + __expf(-g));
      fusedL[m * 768 + c] = (_Float16)(g * cat_val(row0 + m, c));
    }
  }
  __syncthreads();

  // out = fused(16x768) @ ow(768x256): 16 N-tiles, 2 per wave
  v8f acco[2];
  acco[0] = (v8f){}; acco[1] = (v8f){};
  for (int k0 = 0; k0 < 768; k0 += 32) {
    v16h af = load_a_frag(fusedL + k0, 768, lane);
#pragma unroll
    for (int i = 0; i < 2; ++i) {
      const int nt = wv * 2 + i;
      v16h bf = load_b_frag(owh + (size_t)k0 * 256 + nt * 16, 256, lane);
      acco[i] = __builtin_amdgcn_wmma_f32_16x16x32_f16(
          false, af, false, bf, (short)0, acco[i], false, false);
    }
  }
#pragma unroll
  for (int i = 0; i < 2; ++i) {
    const int nt = wv * 2 + i;
#pragma unroll
    for (int v = 0; v < 8; ++v)
      outL[CD_ROW(v, lane) * 256 + nt * 16 + CD_COL(lane)] =
          acco[i][v] + ob[nt * 16 + CD_COL(lane)];
  }
  __syncthreads();

  // LayerNorm + ReLU per row (16 threads per row)
  const int rr = tid >> 4, cg = tid & 15;
  float ps = 0.f, pq = 0.f;
  for (int c = cg * 16; c < cg * 16 + 16; ++c) {
    const float v = outL[rr * 256 + c];
    ps += v; pq += v * v;
  }
  red[tid] = ps;
  red[256 + tid] = pq;
  __syncthreads();
  float s = 0.f, q = 0.f;
  for (int k = 0; k < 16; ++k) { s += red[rr * 16 + k]; q += red[256 + rr * 16 + k]; }
  const float mu  = s * (1.f / 256.f);
  const float var = q * (1.f / 256.f) - mu * mu;
  const float inv = rsqrtf(var + GAT_EPS);
  const int bt = row0 + rr;
  for (int c = cg * 16; c < cg * 16 + 16; ++c) {
    float v = (outL[rr * 256 + c] - mu) * inv * og[c] + obeta[c];
    out[(size_t)bt * 256 + c] = v > 0.f ? v : 0.f;
  }
}

// ---------------------------------------------------------------------------

extern "C" void kernel_launch(void* const* d_in, const int* in_sizes, int n_in,
                              void* d_out, int out_size, void* d_ws, size_t ws_size,
                              hipStream_t stream) {
  (void)in_sizes; (void)n_in; (void)out_size; (void)ws_size;

  const float* x = (const float*)d_in[0];
  const float* adj[3] = {(const float*)d_in[1], (const float*)d_in[2],
                         (const float*)d_in[3]};
  const int Ns[3]   = {42, 40, 11};
  const int Nps[3]  = {48, 48, 16};
  const int xoff[3] = {0, 126, 246};
  const int SB[3]   = {4, 16, 28};  // per-stream param base index

  const float* gate_w = (const float*)d_in[40];
  const float* gate_b = (const float*)d_in[41];
  const float* ow     = (const float*)d_in[42];
  const float* ob     = (const float*)d_in[43];
  const float* og     = (const float*)d_in[44];
  const float* obeta  = (const float*)d_in[45];

  // ---- workspace carving ----
  char* ws = (char*)d_ws;
  size_t off = 0;
  auto walloc = [&](size_t bytes) -> char* {
    char* p = ws + off;
    off = (off + bytes + 255) & ~(size_t)255;
    return p;
  };

  _Float16* h1h[3];  float* h2f[3];  float* stats[3];  float* sout[3];
  _Float16* W2h[3];
  for (int s = 0; s < 3; ++s) {
    h1h[s]   = (_Float16*)walloc((size_t)BT_TOT * Ns[s] * 256 * sizeof(_Float16));
    h2f[s]   = (float*)walloc((size_t)BT_TOT * Ns[s] * 64 * sizeof(float));
    stats[s] = (float*)walloc(1280 * sizeof(float));
    // layout: sum1[0..255] sq1[256..511] sum2[512..575] sq2[576..639]
    //         scale1[640..] bias1[896..] scale2[1152..] bias2[1216..]
    sout[s]  = (float*)walloc((size_t)BT_TOT * 256 * sizeof(float));
    W2h[s]   = (_Float16*)walloc((size_t)256 * 256 * sizeof(_Float16));
  }
  _Float16* gwh = (_Float16*)walloc((size_t)768 * 768 * sizeof(_Float16));
  _Float16* owh = (_Float16*)walloc((size_t)768 * 256 * sizeof(_Float16));

  const int SM1 = NMAX * 256 * 4 + 4 * NMAX * NMAX * 4 + 4 * NMAX * 4 * 2 + NMAX * 3 * 4;
  const int SM3 = NMAX * 256 * 4 + 4 * NMAX * NMAX * 4 + 4 * NMAX * 4 * 2 +
                  NMAX * 256 * 2 + 256 * 256 * 2;   // + 128KB LDS-resident W2
  const int SM6 = 16 * 768 * 2 * 2 + 16 * 256 * 4 + 512 * 4;

  // ---- K0: zero stats, convert weights to f16 ----
  for (int s = 0; s < 3; ++s)
    zero_f32_kernel<<<3, 256, 0, stream>>>(stats[s], 640);
  for (int s = 0; s < 3; ++s)
    cvt_f32_to_f16_kernel<<<(65536 + 255) / 256, 256, 0, stream>>>(
        (const float*)d_in[SB[s] + 4], W2h[s], 65536);
  cvt_f32_to_f16_kernel<<<(589824 + 255) / 256, 256, 0, stream>>>(gate_w, gwh, 589824);
  cvt_f32_to_f16_kernel<<<(196608 + 255) / 256, 256, 0, stream>>>(ow, owh, 196608);

  // ---- per-stream pipeline ----
  for (int s = 0; s < 3; ++s) {
    const int base = SB[s];
    const float* W1 = (const float*)d_in[base + 0];
    const float* a1 = (const float*)d_in[base + 1];
    const float* g1 = (const float*)d_in[base + 2];
    const float* b1 = (const float*)d_in[base + 3];
    const float* a2 = (const float*)d_in[base + 5];
    const float* g2 = (const float*)d_in[base + 6];
    const float* b2 = (const float*)d_in[base + 7];
    const float* pw = (const float*)d_in[base + 8];
    const float* pb = (const float*)d_in[base + 9];
    const float* lg = (const float*)d_in[base + 10];
    const float* lb = (const float*)d_in[base + 11];

    float* sum1 = stats[s] + 0,    *sq1 = stats[s] + 256;
    float* sum2 = stats[s] + 512,  *sq2 = stats[s] + 576;
    float* sc1  = stats[s] + 640,  *bi1 = stats[s] + 896;
    float* sc2  = stats[s] + 1152, *bi2 = stats[s] + 1216;

    const float invM = 1.f / (float)((size_t)BT_TOT * Ns[s]);

    gat1_kernel<<<BT_TOT, 256, SM1, stream>>>(
        x, xoff[s], 279, adj[s], Ns[s], W1, a1, h1h[s], sum1, sq1);
    bn_finalize_kernel<<<1, 256, 0, stream>>>(sum1, sq1, g1, b1, sc1, bi1, 256, invM);
    gat2_kernel<<<BT_TOT, 256, SM3, stream>>>(
        h1h[s], sc1, bi1, adj[s], Ns[s], Nps[s], W2h[s], a2, h2f[s], sum2, sq2);
    bn_finalize_kernel<<<1, 64, 0, stream>>>(sum2, sq2, g2, b2, sc2, bi2, 64, invM);
    pool_kernel<<<BT_TOT, 256, 0, stream>>>(
        h2f[s], Ns[s], sc2, bi2, pw, pb, lg, lb, sout[s]);
  }

  // ---- fusion ----
  fusion_kernel<<<BT_TOT / 16, 256, SM6, stream>>>(
      sout[0], sout[1], sout[2], gwh, gate_b, owh, ob, og, obeta,
      (float*)d_out);
}
